// MVPAnchorFormer_67525475828183
// MI455X (gfx1250) — compile-verified, matrix operand order checked by reading, not run on hardware
//
#include <hip/hip_runtime.h>
#include <hip/hip_bf16.h>
#include <math.h>

typedef __bf16 bf16;
typedef __bf16 v16bf __attribute__((ext_vector_type(16)));
typedef float  v8f   __attribute__((ext_vector_type(8)));

namespace {
constexpr int kBT = 32, kS = 577, kD = 1024, kHC = 16, kT = 4, kTN = 256;
constexpr int kHEADS = 8, kDHEAD = 64, kINNER = 512, kDEPTH = 6, kGPAH = 512, kLM = 4096;
constexpr int kMLAT = kBT * (kTN + 1);   // 8224 latent rows
constexpr int kMX   = kBT * kS;          // 18464 context rows
constexpr int kSP   = kS - 1;            // 576 patches
}

__device__ __forceinline__ v8f zero8() {
  v8f v;
#pragma unroll
  for (int i = 0; i < 8; ++i) v[i] = 0.f;
  return v;
}

__device__ __forceinline__ v8f wmma_bf16(v16bf a, v16bf b, v8f c) {
  return __builtin_amdgcn_wmma_f32_16x16x32_bf16(false, a, false, b, (short)0, c,
                                                 false, false);
}

// CDNA5 async global->LDS copy (16B per lane), tracked with ASYNCcnt.
__device__ __forceinline__ void async_ld16(unsigned lds_off, const bf16* gptr) {
  asm volatile("global_load_async_to_lds_b128 %0, %1, off"
               :: "v"(lds_off), "v"(gptr) : "memory");
}

// ds_swizzle_b32 XOR butterfly (group-of-32 mode: and=0x1f, or=0, xor=XM)
template <int XM>
__device__ __forceinline__ float swz_xor(float x) {
  return __int_as_float(
      __builtin_amdgcn_ds_swizzle(__float_as_int(x), 0x1f | (XM << 10)));
}
__device__ __forceinline__ float red16_max(float x) {
  x = fmaxf(x, swz_xor<1>(x));
  x = fmaxf(x, swz_xor<2>(x));
  x = fmaxf(x, swz_xor<4>(x));
  x = fmaxf(x, swz_xor<8>(x));
  return x;
}
__device__ __forceinline__ float red16_sum(float x) {
  x += swz_xor<1>(x);
  x += swz_xor<2>(x);
  x += swz_xor<4>(x);
  x += swz_xor<8>(x);
  return x;
}

// ---------------------------------------------------------------------------
// Anchor selection: per-batch-item block. Replicates top_k-per-head + mean
// fallback + dedupe + sort from the reference (final result is a sorted set).
// ---------------------------------------------------------------------------
__device__ __forceinline__ int block_argmax(const float* vals, int n, float* rv,
                                            int* ri, int tid) {
  float bv = -1e30f; int bi = 0x7fffffff;
  for (int i = tid; i < n; i += 256) {
    float v = vals[i];
    if (v > bv || (v == bv && i < bi)) { bv = v; bi = i; }
  }
  rv[tid] = bv; ri[tid] = bi; __syncthreads();
  for (int st = 128; st > 0; st >>= 1) {
    if (tid < st) {
      if (rv[tid + st] > rv[tid] ||
          (rv[tid + st] == rv[tid] && ri[tid + st] < ri[tid])) {
        rv[tid] = rv[tid + st]; ri[tid] = ri[tid + st];
      }
    }
    __syncthreads();
  }
  int w = ri[0]; __syncthreads();
  return w;
}

__global__ __launch_bounds__(256)
void anchors_kernel(const float* __restrict__ attn, int* __restrict__ take) {
  const int b = blockIdx.x, tid = threadIdx.x;
  __shared__ float sc[kSP];
  __shared__ float smean[kSP];
  __shared__ int   flag[kS];
  __shared__ float rv[256];
  __shared__ int   ri[256];

  for (int i = tid; i < kS; i += 256) flag[i] = 0;
  for (int i = tid; i < kSP; i += 256) smean[i] = 0.f;
  __syncthreads();

  for (int h = 0; h < kHC; ++h) {
    const float* src = attn + ((size_t)(b * kHC + h) * kS) * kS + 1;  // [b,h,0,1:]
    for (int i = tid; i < kSP; i += 256) { float v = src[i]; sc[i] = v; smean[i] += v; }
    __syncthreads();
    for (int t = 0; t < 16; ++t) {            // per_head = 16 candidates
      int w = block_argmax(sc, kSP, rv, ri, tid);
      if (tid == 0) { flag[w + 1] = 1; sc[w] = -1e30f; }
      __syncthreads();
    }
  }
  // unique count
  int c = 0;
  for (int i = tid; i < kSP; i += 256) c += flag[i + 1];
  ri[tid] = c; __syncthreads();
  for (int st = 128; st > 0; st >>= 1) { if (tid < st) ri[tid] += ri[tid + st]; __syncthreads(); }
  int need = kTN - ri[0]; if (need < 0) need = 0;
  __syncthreads();
  // mask already-selected in mean scores, fill remainder in mean order
  for (int i = tid; i < kSP; i += 256) if (flag[i + 1]) smean[i] = -1e30f;
  __syncthreads();
  for (int r = 0; r < need; ++r) {
    int w = block_argmax(smean, kSP, rv, ri, tid);
    if (tid == 0) { flag[w + 1] = 1; smean[w] = -1e30f; }
    __syncthreads();
  }
  if (tid == 0) {
    int* o = take + b * (kTN + 1);
    int cnt = 0;
    o[cnt++] = 0;
    for (int i = 1; i < kS && cnt < kTN + 1; ++i) if (flag[i]) o[cnt++] = i;
    while (cnt < kTN + 1) o[cnt++] = 0;
  }
}

__global__ __launch_bounds__(256)
void gather_kernel(const float* __restrict__ Rv, const int* __restrict__ take,
                   float* __restrict__ lat) {
  const int row = blockIdx.x;            // b*257 + j
  const int b = row / (kTN + 1);
  const int s = take[row];
  const float* src = Rv + ((size_t)b * kS + s) * kD;
  float* dst = lat + (size_t)row * kD;
  for (int i = threadIdx.x; i < kD; i += 256) dst[i] = src[i];
}

// ---------------------------------------------------------------------------
// LayerNorm -> bf16 (optionally also fp32)
// ---------------------------------------------------------------------------
template <bool WF>
__global__ __launch_bounds__(256)
void ln_kernel(const float* __restrict__ X, const float* __restrict__ g,
               const float* __restrict__ bb, bf16* __restrict__ Ob,
               float* __restrict__ Of) {
  const int row = blockIdx.x, tid = threadIdx.x;
  const float* x = X + (size_t)row * kD;
  __shared__ float red[256];
  float s = 0.f;
  for (int i = tid; i < kD; i += 256) s += x[i];
  red[tid] = s; __syncthreads();
  for (int st = 128; st > 0; st >>= 1) { if (tid < st) red[tid] += red[tid + st]; __syncthreads(); }
  const float mean = red[0] * (1.f / kD);
  __syncthreads();
  float v = 0.f;
  for (int i = tid; i < kD; i += 256) { float d = x[i] - mean; v += d * d; }
  red[tid] = v; __syncthreads();
  for (int st = 128; st > 0; st >>= 1) { if (tid < st) red[tid] += red[tid + st]; __syncthreads(); }
  const float rstd = rsqrtf(red[0] * (1.f / kD) + 1e-5f);
  for (int i = tid; i < kD; i += 256) {
    float y = (x[i] - mean) * rstd * g[i] + bb[i];
    Ob[(size_t)row * kD + i] = (bf16)y;
    if (WF) Of[(size_t)row * kD + i] = y;
  }
}

__global__ void cvt_kernel(const float* __restrict__ x, bf16* __restrict__ y, int n) {
  for (int i = blockIdx.x * blockDim.x + threadIdx.x; i < n;
       i += gridDim.x * blockDim.x)
    y[i] = (bf16)x[i];
}

// ---------------------------------------------------------------------------
// Tiled WMMA bf16 GEMM: C = epilogue(A[MxK] @ B[KxN]).
// Block tile 128x64, 8 waves each owning a 32x32 tile (4 accumulators,
// 4 v_wmma per K-step). A/B tiles staged with double-buffered
// GLOBAL_LOAD_ASYNC_TO_LDS_B128 (ASYNCcnt-pipelined against the WMMAs).
// ---------------------------------------------------------------------------
template <bool GELU_, bool RES, bool OBF, bool BIAS>
__global__ __launch_bounds__(256)
void gemm_wmma(const bf16* __restrict__ A, const bf16* __restrict__ B,
               float* __restrict__ Cf, bf16* __restrict__ Cb,
               const float* __restrict__ bias, const float* __restrict__ resid,
               int M, int N, int K, float scale) {
  __shared__ __align__(16) bf16 sA[2][128][40];   // 80B row stride (16B aligned)
  __shared__ __align__(16) bf16 sB[2][32][72];    // 144B row stride
  const int tid = threadIdx.x;
  const int lane = tid & 31;
  const int wave = tid >> 5;
  const int wrow = wave >> 1;   // 0..3 -> 32-row strip
  const int wcol = wave & 1;    // 0..1 -> 32-col strip
  const int m0 = blockIdx.y * 128;
  const int n0 = blockIdx.x * 64;
  const int colL = lane & 15;
  const int roff = (lane & 16) ? 8 : 0;
  const int hs = roff;

  v8f acc00 = zero8(), acc01 = zero8(), acc10 = zero8(), acc11 = zero8();

  // staging decomposition: A tile 128x32 -> 512 16B chunks (2/thread),
  // B tile 32x64 -> 256 16B chunks (1/thread)
  const int aRow = tid >> 2, aCol = (tid & 3) * 8;
  const int bRow = tid >> 3, bCol = (tid & 7) * 8;
  int gmA0 = m0 + aRow;       if (gmA0 >= M) gmA0 = M - 1;   // clamp: masked rows
  int gmA1 = m0 + aRow + 64;  if (gmA1 >= M) gmA1 = M - 1;

  auto stage = [&](int buf, int k0) {
    async_ld16((unsigned)(uintptr_t)&sA[buf][aRow][aCol],
               A + (size_t)gmA0 * K + k0 + aCol);
    async_ld16((unsigned)(uintptr_t)&sA[buf][aRow + 64][aCol],
               A + (size_t)gmA1 * K + k0 + aCol);
    async_ld16((unsigned)(uintptr_t)&sB[buf][bRow][bCol],
               B + (size_t)(k0 + bRow) * N + n0 + bCol);
  };

  stage(0, 0);
  int buf = 0;
  for (int k0 = 0; k0 < K; k0 += 32) {
    const bool more = (k0 + 32) < K;
    if (more) {
      stage(buf ^ 1, k0 + 32);
      asm volatile("s_wait_asynccnt 0x3" ::: "memory");  // previous stage done
    } else {
      asm volatile("s_wait_asynccnt 0x0" ::: "memory");
    }
    __syncthreads();

    v16bf aLo, aHi, b0, b1;
    {
      const bf16* arL = &sA[buf][32 * wrow + colL][0];
      const bf16* arH = &sA[buf][32 * wrow + 16 + colL][0];
#pragma unroll
      for (int j = 0; j < 8; ++j) {
        aLo[j] = arL[hs + j]; aLo[8 + j] = arL[16 + hs + j];
        aHi[j] = arH[hs + j]; aHi[8 + j] = arH[16 + hs + j];
      }
    }
    {
      const int c0 = 32 * wcol + colL;
#pragma unroll
      for (int j = 0; j < 8; ++j) {
        b0[j] = sB[buf][hs + j][c0];      b0[8 + j] = sB[buf][16 + hs + j][c0];
        b1[j] = sB[buf][hs + j][c0 + 16]; b1[8 + j] = sB[buf][16 + hs + j][c0 + 16];
      }
    }
    acc00 = wmma_bf16(aLo, b0, acc00);
    acc01 = wmma_bf16(aLo, b1, acc01);
    acc10 = wmma_bf16(aHi, b0, acc10);
    acc11 = wmma_bf16(aHi, b1, acc11);
    __syncthreads();
    buf ^= 1;
  }

#pragma unroll
  for (int mi = 0; mi < 2; ++mi) {
#pragma unroll
    for (int sub = 0; sub < 2; ++sub) {
      v8f acc = (mi == 0) ? (sub == 0 ? acc00 : acc01)
                          : (sub == 0 ? acc10 : acc11);
      const int gcol = n0 + 32 * wcol + 16 * sub + colL;
#pragma unroll
      for (int i = 0; i < 8; ++i) {
        const int grow = m0 + 32 * wrow + 16 * mi + roff + i;
        if (grow < M) {
          float v = acc[i] * scale;
          if (BIAS) v += bias[gcol];
          if (GELU_) v = 0.5f * v * (1.f + erff(v * 0.70710678f));
          if (RES) v += resid[(size_t)grow * N + gcol];
          if (OBF) Cb[(size_t)grow * N + gcol] = (bf16)v;
          else     Cf[(size_t)grow * N + gcol] = v;
        }
      }
    }
  }
}

// ---------------------------------------------------------------------------
// Flash attention: one wave per (b, h, 16-query tile). Online softmax with
// ds_swizzle XOR reductions, WMMA for q@k^T and p@v; V tile DMA'd to LDS with
// async loads (overlapped with the S-tile WMMAs); P staged through LDS to
// convert C-layout -> A-layout.
// ---------------------------------------------------------------------------
__global__ __launch_bounds__(32)
void flash_attn(const bf16* __restrict__ Q, const bf16* __restrict__ KV,
                bf16* __restrict__ O) {
  __shared__ __align__(16) bf16 sP[16][40];
  __shared__ __align__(16) bf16 sV[32][72];
  const int lane = threadIdx.x;
  const int qt = blockIdx.x, h = blockIdx.y, b = blockIdx.z;
  const int m0 = qt * 16;
  const int colL = lane & 15;
  const int roff = (lane & 16) ? 8 : 0;
  const int hs = roff;

  int arow = m0 + colL; if (arow > kTN) arow = kTN;
  const bf16* qrow = Q + ((size_t)(b * (kTN + 1) + arow)) * kINNER + h * kDHEAD;
  v16bf a0, a1;
#pragma unroll
  for (int j = 0; j < 8; ++j) {
    a0[j] = qrow[hs + j];      a0[8 + j] = qrow[16 + hs + j];
    a1[j] = qrow[32 + hs + j]; a1[8 + j] = qrow[48 + hs + j];
  }

  float mrow[8], lrow[8];
  v8f o0 = zero8(), o1 = zero8(), o2 = zero8(), o3 = zero8();
#pragma unroll
  for (int i = 0; i < 8; ++i) { mrow[i] = -1e30f; lrow[i] = 0.f; }

  for (int kb = 0; kb < kS; kb += 32) {
    {  // async-stage V tile (32 keys x 64); overlaps with S computation below
      int key = kb + lane; if (key >= kS) key = kS - 1;
      const bf16* vsrc = KV + ((size_t)(b * kS + key)) * kD + kINNER + h * kDHEAD;
      const unsigned vl = (unsigned)(uintptr_t)&sV[lane][0];
#pragma unroll
      for (int c = 0; c < 8; ++c) async_ld16(vl + c * 16, vsrc + c * 8);
    }
    v8f s0 = zero8(), s1 = zero8();
    {
      int key0 = kb + colL;      int k0c = key0 < kS ? key0 : kS - 1;
      int key1 = kb + 16 + colL; int k1c = key1 < kS ? key1 : kS - 1;
      const bf16* kr0 = KV + ((size_t)(b * kS + k0c)) * kD + h * kDHEAD;
      const bf16* kr1 = KV + ((size_t)(b * kS + k1c)) * kD + h * kDHEAD;
      v16bf bk;
#pragma unroll
      for (int j = 0; j < 8; ++j) { bk[j] = kr0[hs + j]; bk[8 + j] = kr0[16 + hs + j]; }
      s0 = wmma_bf16(a0, bk, s0);
#pragma unroll
      for (int j = 0; j < 8; ++j) { bk[j] = kr0[32 + hs + j]; bk[8 + j] = kr0[48 + hs + j]; }
      s0 = wmma_bf16(a1, bk, s0);
#pragma unroll
      for (int j = 0; j < 8; ++j) { bk[j] = kr1[hs + j]; bk[8 + j] = kr1[16 + hs + j]; }
      s1 = wmma_bf16(a0, bk, s1);
#pragma unroll
      for (int j = 0; j < 8; ++j) { bk[j] = kr1[32 + hs + j]; bk[8 + j] = kr1[48 + hs + j]; }
      s1 = wmma_bf16(a1, bk, s1);
    }
    const bool v0 = (kb + colL) < kS;
    const bool v1 = (kb + 16 + colL) < kS;
#pragma unroll
    for (int i = 0; i < 8; ++i) {
      float sv0 = v0 ? s0[i] : -1e30f;
      float sv1 = v1 ? s1[i] : -1e30f;
      float mx = red16_max(fmaxf(sv0, sv1));
      float mnew = fmaxf(mrow[i], mx);
      float alpha = __expf(mrow[i] - mnew);
      float p0 = v0 ? __expf(sv0 - mnew) : 0.f;
      float p1 = v1 ? __expf(sv1 - mnew) : 0.f;
      float ps = red16_sum(p0 + p1);
      lrow[i] = lrow[i] * alpha + ps;
      mrow[i] = mnew;
      o0[i] *= alpha; o1[i] *= alpha; o2[i] *= alpha; o3[i] *= alpha;
      sP[roff + i][colL]      = (bf16)p0;
      sP[roff + i][16 + colL] = (bf16)p1;
    }
    asm volatile("s_wait_asynccnt 0x0" ::: "memory");  // V tile landed in LDS
    __syncthreads();
    v16bf ap;
    {
      const bf16* pr = &sP[colL][0];
#pragma unroll
      for (int j = 0; j < 8; ++j) { ap[j] = pr[hs + j]; ap[8 + j] = pr[16 + hs + j]; }
    }
#pragma unroll
    for (int g = 0; g < 4; ++g) {
      v16bf bv;
      const int c = g * 16 + colL;
#pragma unroll
      for (int j = 0; j < 8; ++j) { bv[j] = sV[hs + j][c]; bv[8 + j] = sV[16 + hs + j][c]; }
      v8f& og = (g == 0) ? o0 : (g == 1) ? o1 : (g == 2) ? o2 : o3;
      og = wmma_bf16(ap, bv, og);
    }
    __syncthreads();
  }

#pragma unroll
  for (int i = 0; i < 8; ++i) {
    const int r = m0 + roff + i;
    if (r <= kTN) {
      const float inv = 1.f / lrow[i];
      bf16* dst = O + ((size_t)(b * (kTN + 1) + r)) * kINNER + h * kDHEAD;
      dst[colL]      = (bf16)(o0[i] * inv);
      dst[16 + colL] = (bf16)(o1[i] * inv);
      dst[32 + colL] = (bf16)(o2[i] * inv);
      dst[48 + colL] = (bf16)(o3[i] * inv);
    }
  }
}

// ---------------------------------------------------------------------------
// Gated attention pooling (scores, sigmoid gate, softmax, weighted sum)
// ---------------------------------------------------------------------------
__global__ __launch_bounds__(256)
void pool_kernel(const float* __restrict__ X, const bf16* __restrict__ Mid,
                 const float* __restrict__ qv, const float* __restrict__ g2,
                 const float* __restrict__ b2, float* __restrict__ Of,
                 bf16* __restrict__ Ob, int Mc) {
  const int gi = blockIdx.x, tid = threadIdx.x;
  const float* Xg = X + (size_t)gi * Mc * kD;
  const bf16*  Mg = Mid + (size_t)gi * Mc * kGPAH;
  __shared__ float w[288];
  __shared__ float red[256];
  for (int m = tid; m < Mc; m += 256) {
    float sc = 0.f;
    const float* xr = Xg + (size_t)m * kD;
    for (int d = 0; d < kD; ++d) sc += xr[d] * qv[d];
    float gl = 0.f;
    const bf16* mr = Mg + (size_t)m * kGPAH;
    for (int d = 0; d < kGPAH; ++d) gl += (float)mr[d] * g2[d];
    gl += b2[0];
    const float gate = 1.f / (1.f + __expf(-gl));
    w[m] = sc * gate;
  }
  __syncthreads();
  float mx = -1e30f;
  for (int m = tid; m < Mc; m += 256) mx = fmaxf(mx, w[m]);
  red[tid] = mx; __syncthreads();
  for (int st = 128; st > 0; st >>= 1) { if (tid < st) red[tid] = fmaxf(red[tid], red[tid + st]); __syncthreads(); }
  const float wm = red[0]; __syncthreads();
  float sum = 0.f;
  for (int m = tid; m < Mc; m += 256) { float e = __expf(w[m] - wm); w[m] = e; sum += e; }
  red[tid] = sum; __syncthreads();
  for (int st = 128; st > 0; st >>= 1) { if (tid < st) red[tid] += red[tid + st]; __syncthreads(); }
  const float inv = 1.f / red[0];
  __syncthreads();
  for (int d = tid; d < kD; d += 256) {
    float acc = 0.f;
    for (int m = 0; m < Mc; ++m) acc += w[m] * Xg[(size_t)m * kD + d];
    acc *= inv;
    Of[(size_t)gi * kD + d] = acc;
    Ob[(size_t)gi * kD + d] = (bf16)acc;
  }
}

// ---------------------------------------------------------------------------
extern "C" void kernel_launch(void* const* d_in, const int* in_sizes, int n_in,
                              void* d_out, int out_size, void* d_ws, size_t ws_size,
                              hipStream_t stream) {
  (void)in_sizes; (void)n_in; (void)out_size; (void)ws_size;
  const float* Rv     = (const float*)d_in[0];
  const float* attn   = (const float*)d_in[1];
  const float* ln_x_g = (const float*)d_in[2];
  const float* ln_x_b = (const float*)d_in[3];
  const float* ln_l_g = (const float*)d_in[4];
  const float* ln_l_b = (const float*)d_in[5];
  const float* Wq     = (const float*)d_in[6];
  const float* Wkv    = (const float*)d_in[7];
  const float* Wo     = (const float*)d_in[8];
  const float* ff_g   = (const float*)d_in[9];
  const float* ff_b   = (const float*)d_in[10];
  const float* W1     = (const float*)d_in[11];
  const float* W2     = (const float*)d_in[12];
  const float* norm_g = (const float*)d_in[13];
  const float* norm_b = (const float*)d_in[14];
  const float* tok_q  = (const float*)d_in[15];
  const float* tok_g1 = (const float*)d_in[16];
  const float* tok_b1 = (const float*)d_in[17];
  const float* tok_g2 = (const float*)d_in[18];
  const float* tok_b2 = (const float*)d_in[19];
  const float* frm_q  = (const float*)d_in[20];
  const float* frm_g1 = (const float*)d_in[21];
  const float* frm_b1 = (const float*)d_in[22];
  const float* frm_g2 = (const float*)d_in[23];
  const float* frm_b2 = (const float*)d_in[24];
  const float* proj_w = (const float*)d_in[25];
  const float* proj_b = (const float*)d_in[26];

  // workspace carve (256B aligned)
  size_t off = 0;
  auto alloc = [&](size_t bytes) -> void* {
    void* p = (char*)d_ws + off;
    off = (off + bytes + 255) & ~(size_t)255;
    return p;
  };
  int*   take   = (int*)  alloc((size_t)kMLAT * sizeof(int));
  float* lat    = (float*)alloc((size_t)kMLAT * kD * 4);
  bf16*  latln  = (bf16*) alloc((size_t)kMLAT * kD * 2);
  bf16*  xn     = (bf16*) alloc((size_t)kMX * kD * 2);
  bf16*  qb     = (bf16*) alloc((size_t)kMLAT * kINNER * 2);
  bf16*  kvb    = (bf16*) alloc((size_t)kMX * kD * 2);
  bf16*  attout = (bf16*) alloc((size_t)kMLAT * kINNER * 2);
  bf16*  ffmid  = (bf16*) alloc((size_t)kMLAT * 4 * kD * 2);
  bf16*  wq_bf  = (bf16*) alloc((size_t)kD * kINNER * 2);
  bf16*  wkv_bf = (bf16*) alloc((size_t)kD * 2 * kINNER * 2);
  bf16*  wo_bf  = (bf16*) alloc((size_t)kINNER * kD * 2);
  bf16*  w1_bf  = (bf16*) alloc((size_t)kD * 4 * kD * 2);
  bf16*  w2_bf  = (bf16*) alloc((size_t)4 * kD * kD * 2);
  float* tokf   = (float*)alloc((size_t)kBT * kD * 4);
  bf16*  tokb   = (bf16*) alloc((size_t)kBT * kD * 2);
  bf16*  mid2   = (bf16*) alloc((size_t)kBT * kGPAH * 2);
  float* fusedf = (float*)alloc((size_t)(kBT / kT) * kD * 4);
  bf16*  fusedb = (bf16*) alloc((size_t)(kBT / kT) * kD * 2);
  // aliases (buffers free by the time they're reused)
  float* hvf = (float*)xn;   // final LN fp32 output
  bf16*  hvb = latln;        // final LN bf16 output
  bf16*  midpool = qb;       // GPA hidden (8224x512 bf16)
  bf16*  pg1 = wq_bf;        // converted pooling gate W1s
  bf16*  pw  = w1_bf;        // converted proj_w

  auto cvt = [&](const float* src, bf16* dst, int n) {
    int blocks = (n + 255) / 256; if (blocks > 8192) blocks = 8192;
    cvt_kernel<<<blocks, 256, 0, stream>>>(src, dst, n);
  };

  anchors_kernel<<<kBT, 256, 0, stream>>>(attn, take);
  gather_kernel<<<kMLAT, 256, 0, stream>>>(Rv, take, lat);

  const dim3 gLat1024((kD) / 64, (kMLAT + 127) / 128);
  const dim3 gLat512((kINNER) / 64, (kMLAT + 127) / 128);
  const dim3 gLat4096((4 * kD) / 64, (kMLAT + 127) / 128);
  const dim3 gX1024((2 * kINNER) / 64, (kMX + 127) / 128);

  for (int i = 0; i < kDEPTH; ++i) {
    cvt(Wq  + (size_t)i * kD * kINNER,     wq_bf,  kD * kINNER);
    cvt(Wkv + (size_t)i * kD * 2 * kINNER, wkv_bf, kD * 2 * kINNER);
    cvt(Wo  + (size_t)i * kINNER * kD,     wo_bf,  kINNER * kD);
    cvt(W1  + (size_t)i * kD * 4 * kD,     w1_bf,  kD * 4 * kD);
    cvt(W2  + (size_t)i * 4 * kD * kD,     w2_bf,  4 * kD * kD);

    // k,v = LN(Rv) @ Wkv
    ln_kernel<false><<<kMX, 256, 0, stream>>>(Rv, ln_x_g + i * kD, ln_x_b + i * kD, xn, nullptr);
    gemm_wmma<false, false, true, false><<<gX1024, 256, 0, stream>>>(
        xn, wkv_bf, nullptr, kvb, nullptr, nullptr, kMX, 2 * kINNER, kD, 1.f);

    // q = LN(lat) @ Wq * DHEAD^-0.5
    ln_kernel<false><<<kMLAT, 256, 0, stream>>>(lat, ln_l_g + i * kD, ln_l_b + i * kD, latln, nullptr);
    gemm_wmma<false, false, true, false><<<gLat512, 256, 0, stream>>>(
        latln, wq_bf, nullptr, qb, nullptr, nullptr, kMLAT, kINNER, kD, 0.125f);

    flash_attn<<<dim3(17, kHEADS, kBT), 32, 0, stream>>>(qb, kvb, attout);

    // lat += attout @ Wo
    gemm_wmma<false, true, false, false><<<gLat1024, 256, 0, stream>>>(
        attout, wo_bf, lat, nullptr, nullptr, lat, kMLAT, kD, kINNER, 1.f);

    // lat += gelu(LN(lat) @ W1) @ W2
    ln_kernel<false><<<kMLAT, 256, 0, stream>>>(lat, ff_g + i * kD, ff_b + i * kD, latln, nullptr);
    gemm_wmma<true, false, true, false><<<gLat4096, 256, 0, stream>>>(
        latln, w1_bf, nullptr, ffmid, nullptr, nullptr, kMLAT, 4 * kD, kD, 1.f);
    gemm_wmma<false, true, false, false><<<gLat1024, 256, 0, stream>>>(
        ffmid, w2_bf, lat, nullptr, nullptr, lat, kMLAT, kD, 4 * kD, 1.f);
  }

  // Hv = LN(lat)
  ln_kernel<true><<<kMLAT, 256, 0, stream>>>(lat, norm_g, norm_b, hvb, hvf);

  // token-level GPA pool
  cvt(tok_g1, pg1, kD * kGPAH);
  gemm_wmma<true, false, true, true><<<dim3(kGPAH / 64, (kMLAT + 127) / 128), 256, 0, stream>>>(
      hvb, pg1, nullptr, midpool, tok_b1, nullptr, kMLAT, kGPAH, kD, 1.f);
  pool_kernel<<<kBT, 256, 0, stream>>>(hvf, midpool, tok_q, tok_g2, tok_b2, tokf, tokb, kTN + 1);

  // frame-level GPA pool
  cvt(frm_g1, pg1, kD * kGPAH);
  gemm_wmma<true, false, true, true><<<dim3(kGPAH / 64, 1), 256, 0, stream>>>(
      tokb, pg1, nullptr, mid2, frm_b1, nullptr, kBT, kGPAH, kD, 1.f);
  pool_kernel<<<kBT / kT, 256, 0, stream>>>(tokf, mid2, frm_q, frm_g2, frm_b2, fusedf, fusedb, kT);

  // final projection -> d_out (8 x 4096 fp32)
  cvt(proj_w, pw, kD * kLM);
  gemm_wmma<false, false, false, true><<<dim3(kLM / 64, 1), 256, 0, stream>>>(
      fusedb, pw, (float*)d_out, nullptr, proj_b, nullptr, kBT / kT, kLM, kD, 1.f);
}
